// Int8LlamaMLP_4861902979323
// MI455X (gfx1250) — 8-bit matrix operand order repaired and checked by reading, compiled, not run on hardware
//
#include <hip/hip_runtime.h>
#include <hip/hip_bf16.h>


typedef __attribute__((ext_vector_type(8))) int   v8i;
typedef __attribute__((ext_vector_type(4))) int   v4i;
typedef __attribute__((ext_vector_type(4))) float v4f;

constexpr int T = 8192;
constexpr int H = 4096;
constexpr int I = 11008;

constexpr int LDS_STRIDE = 80;

__device__ __forceinline__ int quantize_i8(float x, float inv_scale) {
    float r = rintf(x * inv_scale);
    r = fminf(fmaxf(r, -128.0f), 127.0f);
    return (int)r;
}

__device__ __forceinline__ unsigned lds_off(const void* p) {
    return (unsigned)(unsigned long long)p;
}

__device__ __forceinline__ void async_load_b128(unsigned lds_addr,
                                                const signed char* g) {
    asm volatile("global_load_async_to_lds_b128 %0, %1, off"
                 :: "v"(lds_addr), "v"((unsigned long long)g)
                 : "memory");
}

__device__ __forceinline__ void wait_async0() {
    asm volatile("s_wait_asynccnt 0" ::: "memory");
}

__global__ __launch_bounds__(256)
void quantize_x_kernel(const float* __restrict__ x,
                       const float* __restrict__ scale,
                       signed char* __restrict__ qx, int n4) {
    int idx = blockIdx.x * blockDim.x + threadIdx.x;
    if (idx >= n4) return;
    const float inv = 1.0f / scale[0];
    v4f v = *(const v4f*)(x + (size_t)idx * 4);
    unsigned int p = ((unsigned int)(unsigned char)quantize_i8(v[0], inv))
                   | ((unsigned int)(unsigned char)quantize_i8(v[1], inv) << 8)
                   | ((unsigned int)(unsigned char)quantize_i8(v[2], inv) << 16)
                   | ((unsigned int)(unsigned char)quantize_i8(v[3], inv) << 24);
    *(unsigned int*)(qx + (size_t)idx * 4) = p;
}

__device__ __forceinline__ v8i load_a_frag(const signed char* __restrict__ As,
                                           int mrow, int half) {
    v8i a;
    const signed char* ap = As + mrow * LDS_STRIDE + half * 16;
#pragma unroll
    for (int g = 0; g < 4; ++g) {
        unsigned long long w = *(const unsigned long long*)(ap + (g >> 1) * 32 + (g & 1) * 8);
        a[2 * g]     = (int)(unsigned int)(w & 0xffffffffull);
        a[2 * g + 1] = (int)(unsigned int)(w >> 32);
    }
    return a;
}

__device__ __forceinline__ v8i load_b_frag(const signed char* __restrict__ Bs,
                                           int nrow, int half) {
    const signed char* bp = Bs + nrow * LDS_STRIDE + half * 16;
    v4i lo = *(const v4i*)bp;
    v4i hi = *(const v4i*)(bp + 32);
    v8i b = {lo[0], lo[1], lo[2], lo[3], hi[0], hi[1], hi[2], hi[3]};
    return b;
}

__device__ __forceinline__ v8i wmma_iu8(v8i a, v8i b, v8i c) {
    return __builtin_amdgcn_wmma_i32_16x16x64_iu8(true, a, true, b, c,
                                                  false, false);
}

__global__ __launch_bounds__(256, 1)
void gateup_silu_quant_kernel(const signed char* __restrict__ qx,
                              const signed char* __restrict__ gate_w,
                              const signed char* __restrict__ up_w,
                              signed char* __restrict__ qh,
                              const float* __restrict__ g_in_s,
                              const float* __restrict__ g_w_s,
                              const float* __restrict__ u_in_s,
                              const float* __restrict__ u_w_s,
                              const float* __restrict__ d_in_s) {
    __shared__ __align__(16) signed char As[2][256 * LDS_STRIDE];
    __shared__ __align__(16) signed char Bg[2][64 * LDS_STRIDE];
    __shared__ __align__(16) signed char Bu[2][64 * LDS_STRIDE];

    const int tid   = threadIdx.x;
    const int wave  = tid >> 5;
    const int lane  = tid & 31;
    const int half  = lane >> 4;
    const int nlane = lane & 15;

    const int rowBase = blockIdx.y * 256;
    const int colBase = blockIdx.x * 64;

    const int sR = tid >> 2;
    const int sC = (tid & 3) << 4;

    unsigned ldsA[2][4], ldsG[2], ldsU[2];
#pragma unroll
    for (int buf = 0; buf < 2; ++buf) {
#pragma unroll
        for (int c = 0; c < 4; ++c)
            ldsA[buf][c] = lds_off(&As[buf][(sR + 64 * c) * LDS_STRIDE + sC]);
        ldsG[buf] = lds_off(&Bg[buf][sR * LDS_STRIDE + sC]);
        ldsU[buf] = lds_off(&Bu[buf][sR * LDS_STRIDE + sC]);
    }
    const signed char* gA[4];
#pragma unroll
    for (int c = 0; c < 4; ++c)
        gA[c] = qx + (size_t)(rowBase + sR + 64 * c) * H + sC;
    const signed char* gG = gate_w + (size_t)(colBase + sR) * H + sC;
    const signed char* gU = up_w   + (size_t)(colBase + sR) * H + sC;

    v8i accg[2][4], accu[2][4];
#pragma unroll
    for (int mi = 0; mi < 2; ++mi)
#pragma unroll
        for (int t = 0; t < 4; ++t) {
            accg[mi][t] = (v8i){0,0,0,0,0,0,0,0};
            accu[mi][t] = (v8i){0,0,0,0,0,0,0,0};
        }

    auto stage_async = [&](int k0, int buf) {
#pragma unroll
        for (int c = 0; c < 4; ++c)
            async_load_b128(ldsA[buf][c], gA[c] + k0);
        async_load_b128(ldsG[buf], gG + k0);
        async_load_b128(ldsU[buf], gU + k0);
    };
    auto compute = [&](int buf) {
        v8i a0 = load_a_frag(As[buf], wave * 32 + nlane, half);
        v8i a1 = load_a_frag(As[buf], wave * 32 + 16 + nlane, half);
#pragma unroll
        for (int t = 0; t < 4; ++t) {
            v8i bg = load_b_frag(Bg[buf], t * 16 + nlane, half);
            accg[0][t] = wmma_iu8(a0, bg, accg[0][t]);
            accg[1][t] = wmma_iu8(a1, bg, accg[1][t]);
            v8i bu = load_b_frag(Bu[buf], t * 16 + nlane, half);
            accu[0][t] = wmma_iu8(a0, bu, accu[0][t]);
            accu[1][t] = wmma_iu8(a1, bu, accu[1][t]);
        }
    };

    constexpr int KSTEPS = H / 64;
    stage_async(0, 0);
    wait_async0();
    __syncthreads();
    for (int kk = 0; kk < KSTEPS; ++kk) {
        const int cur = kk & 1;
        if (kk + 1 < KSTEPS) stage_async((kk + 1) * 64, 1 - cur);
        compute(cur);
        wait_async0();
        __syncthreads();
    }

    const float gscale = g_in_s[0] * g_w_s[0];
    const float uscale = u_in_s[0] * u_w_s[0];
    const float dinv   = 1.0f / d_in_s[0];

#pragma unroll
    for (int mi = 0; mi < 2; ++mi)
#pragma unroll
        for (int t = 0; t < 4; ++t)
#pragma unroll
            for (int r = 0; r < 8; ++r) {
                int m = wave * 32 + mi * 16 + half * 8 + r;
                float gf = (float)accg[mi][t][r] * gscale;
                float uf = (float)accu[mi][t][r] * uscale;
                float gh = (float)(_Float16)gf;
                float sil = gh / (1.0f + __expf(-gh));
                float hv = sil * uf;
                qh[(size_t)(rowBase + m) * I + colBase + t * 16 + nlane] =
                    (signed char)quantize_i8(hv, dinv);
            }
}

__global__ __launch_bounds__(256, 1)
void down_kernel(const signed char* __restrict__ qh,
                 const signed char* __restrict__ down_w,
                 float* __restrict__ out,
                 const float* __restrict__ d_in_s,
                 const float* __restrict__ d_w_s) {
    __shared__ __align__(16) signed char As[2][256 * LDS_STRIDE];
    __shared__ __align__(16) signed char Bs[2][64 * LDS_STRIDE];

    const int tid   = threadIdx.x;
    const int wave  = tid >> 5;
    const int lane  = tid & 31;
    const int half  = lane >> 4;
    const int nlane = lane & 15;

    const int rowBase = blockIdx.y * 256;
    const int colBase = blockIdx.x * 64;

    const int sR = tid >> 2;
    const int sC = (tid & 3) << 4;

    unsigned ldsA[2][4], ldsB[2];
#pragma unroll
    for (int buf = 0; buf < 2; ++buf) {
#pragma unroll
        for (int c = 0; c < 4; ++c)
            ldsA[buf][c] = lds_off(&As[buf][(sR + 64 * c) * LDS_STRIDE + sC]);
        ldsB[buf] = lds_off(&Bs[buf][sR * LDS_STRIDE + sC]);
    }
    const signed char* gA[4];
#pragma unroll
    for (int c = 0; c < 4; ++c)
        gA[c] = qh + (size_t)(rowBase + sR + 64 * c) * I + sC;
    const signed char* gB = down_w + (size_t)(colBase + sR) * I + sC;

    v8i acc[2][4];
#pragma unroll
    for (int mi = 0; mi < 2; ++mi)
#pragma unroll
        for (int t = 0; t < 4; ++t) acc[mi][t] = (v8i){0,0,0,0,0,0,0,0};

    auto stage_async = [&](int k0, int buf) {
#pragma unroll
        for (int c = 0; c < 4; ++c)
            async_load_b128(ldsA[buf][c], gA[c] + k0);
        async_load_b128(ldsB[buf], gB + k0);
    };
    auto compute = [&](int buf) {
        v8i a0 = load_a_frag(As[buf], wave * 32 + nlane, half);
        v8i a1 = load_a_frag(As[buf], wave * 32 + 16 + nlane, half);
#pragma unroll
        for (int t = 0; t < 4; ++t) {
            v8i b = load_b_frag(Bs[buf], t * 16 + nlane, half);
            acc[0][t] = wmma_iu8(a0, b, acc[0][t]);
            acc[1][t] = wmma_iu8(a1, b, acc[1][t]);
        }
    };

    constexpr int KSTEPS = I / 64;
    stage_async(0, 0);
    wait_async0();
    __syncthreads();
    for (int kk = 0; kk < KSTEPS; ++kk) {
        const int cur = kk & 1;
        if (kk + 1 < KSTEPS) stage_async((kk + 1) * 64, 1 - cur);
        compute(cur);
        wait_async0();
        __syncthreads();
    }

    const float oscale = d_in_s[0] * d_w_s[0];
#pragma unroll
    for (int mi = 0; mi < 2; ++mi)
#pragma unroll
        for (int t = 0; t < 4; ++t)
#pragma unroll
            for (int r = 0; r < 8; ++r) {
                int m = wave * 32 + mi * 16 + half * 8 + r;
                out[(size_t)(rowBase + m) * H + colBase + t * 16 + nlane] =
                    (float)acc[mi][t][r] * oscale;
            }
}

extern "C" void kernel_launch(void* const* d_in, const int* in_sizes, int n_in,
                              void* d_out, int out_size, void* d_ws, size_t ws_size,
                              hipStream_t stream) {
    const float*       x        = (const float*)d_in[0];
    const signed char* gate_w   = (const signed char*)d_in[1];
    const signed char* up_w     = (const signed char*)d_in[2];
    const signed char* down_w   = (const signed char*)d_in[3];
    const float*       g_in_s   = (const float*)d_in[4];
    const float*       g_w_s    = (const float*)d_in[5];
    const float*       u_in_s   = (const float*)d_in[6];
    const float*       u_w_s    = (const float*)d_in[7];
    const float*       d_in_s   = (const float*)d_in[8];
    const float*       d_w_s    = (const float*)d_in[9];
    float*             out      = (float*)d_out;

    signed char* qx = (signed char*)d_ws;
    signed char* qh = qx + (size_t)T * H;

    {
        int n4 = (T * H) / 4;
        int blocks = (n4 + 255) / 256;
        quantize_x_kernel<<<blocks, 256, 0, stream>>>(x, g_in_s, qx, n4);
    }
    {
        dim3 grid(I / 64, T / 256);
        gateup_silu_quant_kernel<<<grid, 256, 0, stream>>>(
            qx, gate_w, up_w, qh, g_in_s, g_w_s, u_in_s, u_w_s, d_in_s);
    }
    {
        dim3 grid(H / 64, T / 256);
        down_kernel<<<grid, 256, 0, stream>>>(qh, down_w, out, d_in_s, d_w_s);
    }
}
